// DecoderRNN_65403761983700
// MI455X (gfx1250) — matrix operand order repaired and checked by reading, not compiled, not run on hardware
//
#include <hip/hip_runtime.h>

// ---------------------------------------------------------------------------
// Problem constants (Show-Attend-Tell LSTM decoder)
// ---------------------------------------------------------------------------
#define BB   64      // batch
#define PP   49      // pixels
#define TT   20      // caption length
#define VV   32000   // vocab
#define EE   512     // embed dim
#define DD   512     // hidden dim
#define AD   512     // attention dim
#define ENCD 2048    // encoder dim
#define KCAT 3072    // ctx(2048) + emb(512) + h(512)
#define GG   2048    // 4*D gates

typedef __attribute__((ext_vector_type(16))) __bf16 v16bf;
typedef __attribute__((ext_vector_type(8)))  __bf16 v8bf;
typedef __attribute__((ext_vector_type(8)))  float  v8f;

static __device__ __forceinline__ __bf16 f2bf(float f) {
  union { float f; unsigned u; } v; v.f = f;
  unsigned r = v.u + 0x7FFFu + ((v.u >> 16) & 1u);   // round-to-nearest-even
  union { unsigned short u; __bf16 b; } cv; cv.u = (unsigned short)(r >> 16);
  return cv.b;
}

// ---------------------------------------------------------------------------
// WMMA GEMM, 64x32 per wave (4 m-tiles x 2 n-tiles):
//   C[M,N](f32) = A_bf16[M,K] (row-major, lda) x B_packed + bias
// Each wave loads 2 B fragments per k-tile and reuses each across 4 WMMAs;
// each A fragment is reused across 2 WMMAs. 12 b128 loads : 8 WMMAs per k-tile.
// B_packed layout: element ((nt*kTiles + kt)*32 + lane)*16 + i  holds
//   B[k = kt*32 + (lane>>4)*16 + i][n = nt*16 + (lane&15)]
// A fragment (16x32 bf16): element i of lane L is
//   k = (L>>4)*16 + i, row m = L&15  -> one contiguous 16-elem run (matches B).
// Block = 256 threads = 8 waves; wave w -> n-tiles {2*(blockIdx.x*8+w), +1},
// m0 = blockIdx.y*64.
// ---------------------------------------------------------------------------
__global__ __launch_bounds__(256) void wmma_gemm_bf16(
    const __bf16* __restrict__ Amat, int lda,
    const __bf16* __restrict__ Bp,
    const float* __restrict__ bias,
    float* __restrict__ C, long long ldc, int kTiles)
{
  const int lane = threadIdx.x & 31;
  const int wav  = threadIdx.x >> 5;
  const int nt   = (blockIdx.x * 8 + wav) * 2;
  const int m0   = blockIdx.y * 64;

  const int khalf = (lane >> 4) * 16;
  const __bf16* ap0 = Amat + (long long)(m0 +      (lane & 15)) * lda;
  const __bf16* ap1 = Amat + (long long)(m0 + 16 + (lane & 15)) * lda;
  const __bf16* ap2 = Amat + (long long)(m0 + 32 + (lane & 15)) * lda;
  const __bf16* ap3 = Amat + (long long)(m0 + 48 + (lane & 15)) * lda;
  const long long bstride = (long long)kTiles * 32 * 16;   // one n-tile of B
  const __bf16* bp0 = Bp + (long long)nt * bstride + (long long)lane * 16;
  const __bf16* bp1 = bp0 + bstride;

  v8f acc00 = {}, acc10 = {}, acc20 = {}, acc30 = {};
  v8f acc01 = {}, acc11 = {}, acc21 = {}, acc31 = {};

  for (int kt = 0; kt < kTiles; ++kt) {
    const int k0 = kt * 32 + khalf;

    v16bf b0 = *(const v16bf*)bp0;
    v16bf b1 = *(const v16bf*)bp1;
    bp0 += 32 * 16;
    bp1 += 32 * 16;
    __builtin_prefetch(bp0 + 4 * 32 * 16, 0, 1);   // next B tiles -> cache
    __builtin_prefetch(bp1 + 4 * 32 * 16, 0, 1);

    v8bf a00 = *(const v8bf*)(ap0 + k0);
    v8bf a01 = *(const v8bf*)(ap0 + k0 + 8);
    v8bf a10 = *(const v8bf*)(ap1 + k0);
    v8bf a11 = *(const v8bf*)(ap1 + k0 + 8);
    v8bf a20 = *(const v8bf*)(ap2 + k0);
    v8bf a21 = *(const v8bf*)(ap2 + k0 + 8);
    v8bf a30 = *(const v8bf*)(ap3 + k0);
    v8bf a31 = *(const v8bf*)(ap3 + k0 + 8);

    v16bf a;
#pragma unroll
    for (int i = 0; i < 8; ++i) { a[i] = a00[i]; a[i + 8] = a01[i]; }
    acc00 = __builtin_amdgcn_wmma_f32_16x16x32_bf16(false, a, false, b0, (short)0, acc00, false, false);
    acc01 = __builtin_amdgcn_wmma_f32_16x16x32_bf16(false, a, false, b1, (short)0, acc01, false, false);
#pragma unroll
    for (int i = 0; i < 8; ++i) { a[i] = a10[i]; a[i + 8] = a11[i]; }
    acc10 = __builtin_amdgcn_wmma_f32_16x16x32_bf16(false, a, false, b0, (short)0, acc10, false, false);
    acc11 = __builtin_amdgcn_wmma_f32_16x16x32_bf16(false, a, false, b1, (short)0, acc11, false, false);
#pragma unroll
    for (int i = 0; i < 8; ++i) { a[i] = a20[i]; a[i + 8] = a21[i]; }
    acc20 = __builtin_amdgcn_wmma_f32_16x16x32_bf16(false, a, false, b0, (short)0, acc20, false, false);
    acc21 = __builtin_amdgcn_wmma_f32_16x16x32_bf16(false, a, false, b1, (short)0, acc21, false, false);
#pragma unroll
    for (int i = 0; i < 8; ++i) { a[i] = a30[i]; a[i + 8] = a31[i]; }
    acc30 = __builtin_amdgcn_wmma_f32_16x16x32_bf16(false, a, false, b0, (short)0, acc30, false, false);
    acc31 = __builtin_amdgcn_wmma_f32_16x16x32_bf16(false, a, false, b1, (short)0, acc31, false, false);
  }

  const int   n0 = nt * 16 + (lane & 15);
  const int   n1 = n0 + 16;
  const float bv0 = bias ? bias[n0] : 0.0f;
  const float bv1 = bias ? bias[n1] : 0.0f;
  const int   r0 = m0 + (lane >> 4) * 8;
#pragma unroll
  for (int v = 0; v < 8; ++v) {
    C[(long long)(r0 +      v) * ldc + n0] = acc00[v] + bv0;
    C[(long long)(r0 + 16 + v) * ldc + n0] = acc10[v] + bv0;
    C[(long long)(r0 + 32 + v) * ldc + n0] = acc20[v] + bv0;
    C[(long long)(r0 + 48 + v) * ldc + n0] = acc30[v] + bv0;
    C[(long long)(r0 +      v) * ldc + n1] = acc01[v] + bv1;
    C[(long long)(r0 + 16 + v) * ldc + n1] = acc11[v] + bv1;
    C[(long long)(r0 + 32 + v) * ldc + n1] = acc21[v] + bv1;
    C[(long long)(r0 + 48 + v) * ldc + n1] = acc31[v] + bv1;
  }
}

// ---------------------------------------------------------------------------
// Weight repack into WMMA B-fragment layout (bf16).
// transposed==0 : B[k][n] = W[k*N + n]   (W stored [K,N]: W_out, W_*_att)
// transposed==1 : B[k][n] = W[n*K + k]   (W stored [N,K])
// ---------------------------------------------------------------------------
__global__ __launch_bounds__(256) void pack_b(
    const float* __restrict__ W, __bf16* __restrict__ out,
    int K, int N, int transposed)
{
  long long idx = (long long)blockIdx.x * 256 + threadIdx.x;
  long long total = (long long)K * N;
  if (idx >= total) return;
  const int kTiles = K >> 5;
  int i    = (int)(idx & 15);
  int lane = (int)((idx >> 4) & 31);
  long long rest = idx >> 9;
  int kt = (int)(rest % kTiles);
  int nt = (int)(rest / kTiles);
  int k = kt * 32 + ((lane >> 4) << 4) + i;
  int n = nt * 16 + (lane & 15);
  float v = transposed ? W[(long long)n * K + k] : W[(long long)k * N + n];
  out[idx] = f2bf(v);
}

// Packed B for gates GEMM: K = 3072 (x = [ctx|emb|h]), N = 2048.
// k < 2560: W_ih[n*2560 + k]   (gates = x @ W_ih.T)
// k >=2560: W_hh[n*512 + (k-2560)]
__global__ __launch_bounds__(256) void pack_gates(
    const float* __restrict__ W_ih, const float* __restrict__ W_hh,
    __bf16* __restrict__ out)
{
  long long idx = (long long)blockIdx.x * 256 + threadIdx.x;
  const long long total = (long long)KCAT * GG;
  if (idx >= total) return;
  const int kTiles = KCAT >> 5;
  int i    = (int)(idx & 15);
  int lane = (int)((idx >> 4) & 31);
  long long rest = idx >> 9;
  int kt = (int)(rest % kTiles);
  int nt = (int)(rest / kTiles);
  int k = kt * 32 + ((lane >> 4) << 4) + i;
  int n = nt * 16 + (lane & 15);
  float v = (k < ENCD + EE) ? W_ih[(long long)n * (ENCD + EE) + k]
                            : W_hh[(long long)n * DD + (k - (ENCD + EE))];
  out[idx] = f2bf(v);
}

__global__ __launch_bounds__(256) void bias_combine(
    const float* __restrict__ b_ih, const float* __restrict__ b_hh,
    float* __restrict__ bias_g)
{
  int j = blockIdx.x * 256 + threadIdx.x;
  if (j < GG) bias_g[j] = b_ih[j] + b_hh[j];
}

__global__ __launch_bounds__(256) void convert_features(
    const float* __restrict__ f, __bf16* __restrict__ out)
{
  long long idx = (long long)blockIdx.x * 256 + threadIdx.x;
  if (idx < (long long)BB * PP * ENCD) out[idx] = f2bf(f[idx]);
}

// emb_bf[b, t, j] (t in [0, T-2]) = embed_table[captions[b, t], j]
__global__ __launch_bounds__(256) void embed_lookup(
    const int* __restrict__ captions, const float* __restrict__ table,
    __bf16* __restrict__ emb)
{
  long long idx = (long long)blockIdx.x * 256 + threadIdx.x;
  const long long total = (long long)BB * (TT - 1) * EE;
  if (idx >= total) return;
  int b = (int)(idx / ((TT - 1) * EE));
  int r = (int)(idx % ((TT - 1) * EE));
  int t = r / EE, j = r % EE;
  int tok = captions[b * TT + t];
  emb[idx] = f2bf(table[(long long)tok * EE + j]);
}

// Per-batch mean of features, then h0/c0 GEMVs. h0 -> bf16 slice of A_cat,
// c0 -> f32 state buffer. One block per batch row.
__global__ __launch_bounds__(256) void init_state(
    const float* __restrict__ features,
    const float* __restrict__ W_init_h, const float* __restrict__ b_init_h,
    const float* __restrict__ W_init_c, const float* __restrict__ b_init_c,
    __bf16* __restrict__ A_cat, float* __restrict__ c_state)
{
  __shared__ float s_mean[ENCD];
  const int b = blockIdx.x, tid = threadIdx.x;
  const float* fb = features + (long long)b * PP * ENCD;
  for (int e = tid; e < ENCD; e += 256) {
    float s = 0.f;
    for (int p = 0; p < PP; ++p) s += fb[p * ENCD + e];
    s_mean[e] = s * (1.0f / PP);
  }
  __syncthreads();
  for (int d = tid; d < DD; d += 256) {
    float ah = 0.f, ac = 0.f;
    for (int k = 0; k < ENCD; ++k) {
      float m = s_mean[k];
      ah += m * W_init_h[k * DD + d];
      ac += m * W_init_c[k * DD + d];
    }
    A_cat[b * KCAT + (ENCD + EE) + d] = f2bf(ah + b_init_h[d]);
    c_state[b * DD + d] = ac + b_init_c[d];
  }
}

// Attention for one timestep: scores -> softmax -> ctx; writes ctx (bf16) into
// A_cat[:,0:2048] and copies emb_t (bf16) into A_cat[:,2048:2560].
__global__ __launch_bounds__(256) void attention_step(
    const float* __restrict__ enc_states, const float* __restrict__ dec,
    const float* __restrict__ features,
    const float* __restrict__ w_A, const float* __restrict__ b_A,
    const __bf16* __restrict__ emb, int t, __bf16* __restrict__ A_cat)
{
  __shared__ float s_score[PP];
  __shared__ float s_alpha[PP];
  __shared__ float s_inv;
  const int b = blockIdx.x, tid = threadIdx.x;
  const int lane = tid & 31, wav = tid >> 5;
  const float* enc  = enc_states + (long long)b * PP * AD;
  const float* decb = dec + (long long)b * AD;

  for (int p = wav; p < PP; p += 8) {
    float s = 0.f;
    for (int a = lane; a < AD; a += 32)
      s += tanhf(enc[p * AD + a] + decb[a]) * w_A[a];
    for (int off = 16; off > 0; off >>= 1) s += __shfl_xor(s, off, 32);
    if (lane == 0) s_score[p] = s + b_A[0];
  }
  __syncthreads();
  if (tid == 0) {
    float mx = s_score[0];
    for (int p = 1; p < PP; ++p) mx = fmaxf(mx, s_score[p]);
    float sum = 0.f;
    for (int p = 0; p < PP; ++p) {
      float e = __expf(s_score[p] - mx);
      s_alpha[p] = e; sum += e;
    }
    s_inv = 1.0f / sum;
  }
  __syncthreads();
  const float inv = s_inv;
  const float* fb = features + (long long)b * PP * ENCD;
  for (int e = tid; e < ENCD; e += 256) {
    float acc = 0.f;
    for (int p = 0; p < PP; ++p) acc += s_alpha[p] * fb[p * ENCD + e];
    A_cat[b * KCAT + e] = f2bf(acc * inv);
  }
  const __bf16* et = emb + ((long long)b * (TT - 1) + t) * EE;
  for (int j = tid; j < EE; j += 256)
    A_cat[b * KCAT + ENCD + j] = et[j];
}

// LSTM pointwise: gates[64,2048] (i|f|g|o) + c -> new c, new h (bf16 into A_cat)
__global__ __launch_bounds__(256) void lstm_pointwise(
    const float* __restrict__ gates, float* __restrict__ c_state,
    __bf16* __restrict__ A_cat)
{
  int idx = blockIdx.x * 256 + threadIdx.x;
  if (idx >= BB * DD) return;
  int b = idx / DD, d = idx % DD;
  const float* g = gates + (long long)b * GG;
  float gi = 1.0f / (1.0f + __expf(-g[d]));
  float gf = 1.0f / (1.0f + __expf(-g[DD + d]));
  float gg = tanhf(g[2 * DD + d]);
  float go = 1.0f / (1.0f + __expf(-g[3 * DD + d]));
  float c2 = gf * c_state[idx] + gi * gg;
  float h2 = go * tanhf(c2);
  c_state[idx] = c2;
  A_cat[b * KCAT + (ENCD + EE) + d] = f2bf(h2);
}

// out[:, T-1, :] = 0 (never written by the scan)
__global__ __launch_bounds__(256) void zero_last(float* __restrict__ out)
{
  long long idx = (long long)blockIdx.x * 256 + threadIdx.x;
  if (idx >= (long long)BB * VV) return;
  long long b = idx / VV, v = idx % VV;
  out[(b * TT + (TT - 1)) * (long long)VV + v] = 0.0f;
}

// ---------------------------------------------------------------------------
extern "C" void kernel_launch(void* const* d_in, const int* in_sizes, int n_in,
                              void* d_out, int out_size, void* d_ws, size_t ws_size,
                              hipStream_t stream) {
  (void)in_sizes; (void)n_in; (void)out_size; (void)ws_size;
  const float* features   = (const float*)d_in[0];
  const int*   captions   = (const int*)  d_in[1];
  const float* embed_tab  = (const float*)d_in[2];
  const float* W_ih       = (const float*)d_in[3];
  const float* b_ih       = (const float*)d_in[4];
  const float* W_hh       = (const float*)d_in[5];
  const float* b_hh       = (const float*)d_in[6];
  const float* W_init_h   = (const float*)d_in[7];
  const float* b_init_h   = (const float*)d_in[8];
  const float* W_init_c   = (const float*)d_in[9];
  const float* b_init_c   = (const float*)d_in[10];
  const float* W_enc_att  = (const float*)d_in[11];
  const float* b_enc_att  = (const float*)d_in[12];
  const float* W_dec_att  = (const float*)d_in[13];
  const float* b_dec_att  = (const float*)d_in[14];
  const float* w_A        = (const float*)d_in[15];
  const float* b_A        = (const float*)d_in[16];
  const float* W_out      = (const float*)d_in[17];
  const float* b_out      = (const float*)d_in[18];
  float* out = (float*)d_out;

  // Workspace carve-up (all offsets 256B aligned)
  char* ws = (char*)d_ws;
  size_t o = 0;
  auto carve = [&](size_t bytes) -> char* {
    char* p = ws + o; o += (bytes + 255) & ~(size_t)255; return p;
  };
  __bf16* feat_bf    = (__bf16*)carve((size_t)BB * PP * ENCD * 2);
  __bf16* Bg         = (__bf16*)carve((size_t)KCAT * GG * 2);
  __bf16* Bo         = (__bf16*)carve((size_t)DD * VV * 2);
  __bf16* Bd         = (__bf16*)carve((size_t)DD * AD * 2);
  __bf16* Be         = (__bf16*)carve((size_t)ENCD * AD * 2);
  __bf16* emb_bf     = (__bf16*)carve((size_t)BB * (TT - 1) * EE * 2);
  float*  enc_states = (float*) carve((size_t)BB * PP * AD * 4);
  __bf16* A_cat      = (__bf16*)carve((size_t)BB * KCAT * 2);
  float*  gates      = (float*) carve((size_t)BB * GG * 4);
  float*  c_state    = (float*) carve((size_t)BB * DD * 4);
  float*  dec_ws     = (float*) carve((size_t)BB * AD * 4);
  float*  bias_g     = (float*) carve((size_t)GG * 4);

  auto blks = [](long long n) { return (unsigned)((n + 255) / 256); };

  // One-time preprocessing (per call; deterministic, no caching)
  convert_features<<<blks((long long)BB * PP * ENCD), 256, 0, stream>>>(features, feat_bf);
  pack_b<<<blks((long long)DD * VV), 256, 0, stream>>>(W_out, Bo, DD, VV, 0);
  pack_b<<<blks((long long)DD * AD), 256, 0, stream>>>(W_dec_att, Bd, DD, AD, 0);
  pack_b<<<blks((long long)ENCD * AD), 256, 0, stream>>>(W_enc_att, Be, ENCD, AD, 0);
  pack_gates<<<blks((long long)KCAT * GG), 256, 0, stream>>>(W_ih, W_hh, Bg);
  bias_combine<<<blks(GG), 256, 0, stream>>>(b_ih, b_hh, bias_g);
  embed_lookup<<<blks((long long)BB * (TT - 1) * EE), 256, 0, stream>>>(captions, embed_tab, emb_bf);
  init_state<<<BB, 256, 0, stream>>>(features, W_init_h, b_init_h, W_init_c, b_init_c,
                                     A_cat, c_state);
  // enc_states = features @ W_enc_att + b_enc_att   [3136, 512]
  wmma_gemm_bf16<<<dim3(AD / 256, (BB * PP) / 64), 256, 0, stream>>>(
      feat_bf, ENCD, Be, b_enc_att, enc_states, AD, ENCD / 32);
  zero_last<<<blks((long long)BB * VV), 256, 0, stream>>>(out);

  // Sequential decode: 19 steps
  for (int t = 0; t < TT - 1; ++t) {
    // dec = h @ W_dec_att + b_dec_att   [64, 512]
    wmma_gemm_bf16<<<dim3(AD / 256, BB / 64), 256, 0, stream>>>(
        A_cat + (ENCD + EE), KCAT, Bd, b_dec_att, dec_ws, AD, DD / 32);
    // attention: ctx + emb_t -> A_cat
    attention_step<<<BB, 256, 0, stream>>>(enc_states, dec_ws, features,
                                           w_A, b_A, emb_bf, t, A_cat);
    // gates = [ctx|emb|h] @ [W_ih;W_hh]^T + (b_ih + b_hh)   [64, 2048]
    wmma_gemm_bf16<<<dim3(GG / 256, BB / 64), 256, 0, stream>>>(
        A_cat, KCAT, Bg, bias_g, gates, GG, KCAT / 32);
    // LSTM cell update -> c_state, h2 (bf16 into A_cat)
    lstm_pointwise<<<blks(BB * DD), 256, 0, stream>>>(gates, c_state, A_cat);
    // logits = h2 @ W_out + b_out -> out[:, t, :]   (row stride T*V)
    wmma_gemm_bf16<<<dim3(VV / 256, BB / 64), 256, 0, stream>>>(
        A_cat + (ENCD + EE), KCAT, Bo, b_out, out + (long long)t * VV,
        (long long)TT * VV, DD / 32);
  }
}
